// MSAColumnGlobalAttention_30391188586652
// MI455X (gfx1250) — matrix operand order repaired and checked by reading, not compiled, not run on hardware
//
#include <hip/hip_runtime.h>
#include <hip/hip_bf16.h>

#define S_DIM 512
#define I_DIM 384
#define C_DIM 256
#define H_DIM 8
#define D_DIM 32
#define NROW (S_DIM * I_DIM)   // 196608
#define EPS 1e-5f

typedef __attribute__((ext_vector_type(16))) __bf16 v16bf;
typedef __attribute__((ext_vector_type(8)))  float  v8f;

union TileBF {
    v16bf v;
    uint4 q[2];
};

// ---- helpers -------------------------------------------------------------

__device__ __forceinline__ unsigned short f2bf(float f) {
    unsigned int u = __float_as_uint(f);
    u += 0x7FFFu + ((u >> 16) & 1u);   // round-to-nearest-even
    return (unsigned short)(u >> 16);
}

__device__ __forceinline__ float bf2f(unsigned short h) {
    return __uint_as_float(((unsigned int)h) << 16);
}

__device__ __forceinline__ float waveSum32(float v) {
    #pragma unroll
    for (int m = 16; m > 0; m >>= 1) v += __shfl_xor(v, m, 32);
    return v;
}

__device__ __forceinline__ float blockSum256(float v, float* red) {
    int t = threadIdx.x;
    red[t] = v;
    __syncthreads();
    for (int off = 128; off > 0; off >>= 1) {
        if (t < off) red[t] += red[t + off];
        __syncthreads();
    }
    float r = red[0];
    __syncthreads();
    return r;
}

__device__ __forceinline__ float blockMax256(float v, float* red) {
    int t = threadIdx.x;
    red[t] = v;
    __syncthreads();
    for (int off = 128; off > 0; off >>= 1) {
        if (t < off) red[t] = fmaxf(red[t], red[t + off]);
        __syncthreads();
    }
    float r = red[0];
    __syncthreads();
    return r;
}

// ---- k_pack: W[256,256] fp32 -> WMMA B-operand swizzled bf16 -------------
// layout: pack[(kb*256 + n)*32 + kk] = bf16(W[kb*32+kk][n])
// GEMM lane load: column n = lane&15, K{8h..8h+7, 16+8h..16+8h+7} with h=(lane>=16)
__global__ void k_pack(const float* __restrict__ W, unsigned short* __restrict__ pk) {
    int kb = blockIdx.x;      // 0..7
    int n  = threadIdx.x;     // 0..255
    #pragma unroll
    for (int kk = 0; kk < 32; ++kk) {
        pk[(kb * 256 + n) * 32 + kk] = f2bf(W[(kb * 32 + kk) * C_DIM + n]);
    }
}

// ---- k_mean: mean over s of raw x, then LN -> xm_ln[384,256] -------------
__global__ void k_mean(const float* __restrict__ x, const float* __restrict__ lnw,
                       const float* __restrict__ lnb, float* __restrict__ xm_ln) {
    __shared__ float red[256];
    int i = blockIdx.x, c = threadIdx.x;
    float acc = 0.f;
    for (int s = 0; s < S_DIM; ++s)
        acc += x[((size_t)s * I_DIM + i) * C_DIM + c];
    float v = acc * (1.0f / S_DIM);
    float mu = blockSum256(v, red) * (1.0f / C_DIM);
    float d = v - mu;
    float var = blockSum256(d * d, red) * (1.0f / C_DIM);
    float rs = rsqrtf(var + EPS);
    xm_ln[i * C_DIM + c] = d * rs * lnw[c] + lnb[c];
}

// ---- k_qp: Q = xm_ln @ Wq; P[i,h,c] = (1/sqrt(D)) * sum_d Wk[c,hD+d]*Q ---
__global__ void k_qp(const float* __restrict__ xm_ln, const float* __restrict__ Wq,
                     const float* __restrict__ Wk, float* __restrict__ P) {
    __shared__ float xr[256];
    __shared__ float Qs[256];
    int i = blockIdx.x, t = threadIdx.x;
    xr[t] = xm_ln[i * C_DIM + t];
    __syncthreads();
    float q = 0.f;
    for (int c = 0; c < C_DIM; ++c)
        q = fmaf(xr[c], Wq[c * 256 + t], q);
    Qs[t] = q;
    __syncthreads();
    const float scale = 0.17677669529663687f;  // 1/sqrt(32)
    #pragma unroll
    for (int h = 0; h < H_DIM; ++h) {
        float p = 0.f;
        #pragma unroll
        for (int d = 0; d < D_DIM; ++d)
            p = fmaf(Wk[t * 256 + h * D_DIM + d], Qs[h * D_DIM + d], p);
        P[((size_t)i * H_DIM + h) * C_DIM + t] = p * scale;
    }
}

// ---- k_ln_scores: one WAVE per row; all reductions via shfl --------------
// lane owns 8 contiguous channels c = lane*8 + j
__global__ void k_ln_scores(const float* __restrict__ x, const float* __restrict__ mask,
                            const float* __restrict__ lnw, const float* __restrict__ lnb,
                            const float* __restrict__ P, unsigned short* __restrict__ xln,
                            float* __restrict__ scores) {
    const int wave = threadIdx.x >> 5;
    const int lane = threadIdx.x & 31;
    const size_t row = (size_t)blockIdx.x * 8 + wave;
    const int s = (int)(row / I_DIM), i = (int)(row % I_DIM);
    const int c0 = lane * 8;

    const float4* xr = (const float4*)(x + row * C_DIM + c0);
    float4 v0 = xr[0], v1 = xr[1];
    float xv[8] = {v0.x, v0.y, v0.z, v0.w, v1.x, v1.y, v1.z, v1.w};

    float ps = 0.f;
    #pragma unroll
    for (int j = 0; j < 8; ++j) ps += xv[j];
    float mu = waveSum32(ps) * (1.0f / C_DIM);

    float vs = 0.f;
    #pragma unroll
    for (int j = 0; j < 8; ++j) { float d = xv[j] - mu; vs += d * d; }
    float rs = rsqrtf(waveSum32(vs) * (1.0f / C_DIM) + EPS);

    const float4* wr = (const float4*)(lnw + c0);
    const float4* br = (const float4*)(lnb + c0);
    float4 w0 = wr[0], w1 = wr[1], b0 = br[0], b1 = br[1];
    float lw[8] = {w0.x, w0.y, w0.z, w0.w, w1.x, w1.y, w1.z, w1.w};
    float lb[8] = {b0.x, b0.y, b0.z, b0.w, b1.x, b1.y, b1.z, b1.w};

    union { uint4 q; unsigned short h[8]; } st;
    #pragma unroll
    for (int j = 0; j < 8; ++j) {
        float xl = (xv[j] - mu) * rs * lw[j] + lb[j];
        xv[j] = xl;
        st.h[j] = f2bf(xl);
    }
    *(uint4*)(xln + row * C_DIM + c0) = st.q;   // 16B store per lane

    float bias = (1.0f - mask[row]) * -1e9f;    // mask flat over (s,i)
    #pragma unroll
    for (int h = 0; h < H_DIM; ++h) {
        const float4* pr = (const float4*)(P + ((size_t)i * H_DIM + h) * C_DIM + c0);
        float4 p0 = pr[0], p1 = pr[1];
        float dot = xv[0] * p0.x + xv[1] * p0.y + xv[2] * p0.z + xv[3] * p0.w +
                    xv[4] * p1.x + xv[5] * p1.y + xv[6] * p1.z + xv[7] * p1.w;
        dot = waveSum32(dot);
        if (lane == 0) scores[((size_t)s * H_DIM + h) * I_DIM + i] = dot + bias;
    }
}

// ---- k_softmax over s (512) for each (h,i) -------------------------------
__global__ void k_softmax(const float* __restrict__ scores, float* __restrict__ attn) {
    __shared__ float red[256];
    int col = blockIdx.x;             // h*384 + i
    int h = col / I_DIM, i = col % I_DIM;
    int t = threadIdx.x;
    float v0 = scores[((size_t)t * H_DIM + h) * I_DIM + i];
    float v1 = scores[((size_t)(t + 256) * H_DIM + h) * I_DIM + i];
    float mx = blockMax256(fmaxf(v0, v1), red);
    float e0 = __expf(v0 - mx), e1 = __expf(v1 - mx);
    float inv = 1.0f / blockSum256(e0 + e1, red);
    attn[((size_t)t * H_DIM + h) * I_DIM + i] = e0 * inv;
    attn[((size_t)(t + 256) * H_DIM + h) * I_DIM + i] = e1 * inv;
}

// ---- k_accum: A[h,c] = sum_s attn[s,h,i]*xln[s,i,c]; wvec = A @ Wv -------
__global__ void k_accum(const unsigned short* __restrict__ xln, const float* __restrict__ attn,
                        const float* __restrict__ Wv, float* __restrict__ wvec) {
    __shared__ float at[S_DIM * H_DIM];   // 16 KB
    __shared__ float A[H_DIM][C_DIM];     // 8 KB
    int i = blockIdx.x, t = threadIdx.x;
    for (int idx = t; idx < S_DIM * H_DIM; idx += 256) {
        int s = idx >> 3, h = idx & 7;
        at[idx] = attn[((size_t)s * H_DIM + h) * I_DIM + i];
    }
    __syncthreads();
    float a[H_DIM] = {0.f, 0.f, 0.f, 0.f, 0.f, 0.f, 0.f, 0.f};
    for (int s = 0; s < S_DIM; ++s) {
        float xc = bf2f(xln[((size_t)s * I_DIM + i) * C_DIM + t]);
        const float* ar = &at[s * H_DIM];
        #pragma unroll
        for (int h = 0; h < H_DIM; ++h) a[h] = fmaf(ar[h], xc, a[h]);
    }
    #pragma unroll
    for (int h = 0; h < H_DIM; ++h) A[h][t] = a[h];
    __syncthreads();
    int h = t >> 5, dd = t & 31;
    float w = 0.f;
    for (int c = 0; c < C_DIM; ++c)
        w = fmaf(A[h][c], Wv[c * 256 + h * D_DIM + dd], w);
    wvec[(size_t)i * 256 + t] = w;
}

// ---- k_out: fused  x_ln@Wg -> sigmoid*gate -> @Wo + bo, * mask -----------
// 16-row tiles (same s per tile since 384 % 16 == 0). 8 waves x 2 n-tiles.
// A tile staged LDS-side once via CDNA5 async load (ASYNCcnt), not 8x via L0.
#define LDS_PITCH 272
__launch_bounds__(256)
__global__ void k_out(const unsigned short* __restrict__ xln,
                      const unsigned short* __restrict__ WgP,
                      const unsigned short* __restrict__ WoP,
                      const float* __restrict__ bg, const float* __restrict__ bo,
                      const float* __restrict__ wvec, const float* __restrict__ mask,
                      float* __restrict__ out) {
    __shared__ unsigned short atile[16 * 256];         // 8192 B, pitch 256 halves
    __shared__ unsigned short gtile[16 * LDS_PITCH];   // 8704 B
    const int tid = threadIdx.x;
    const int rowBase = blockIdx.x * 16;
    const int s  = rowBase / I_DIM;
    const int i0 = rowBase % I_DIM;
    const int lane = tid & 31;
    const int wave = tid >> 5;
    const int hi = lane >> 4;        // half-wave select
    const int ln = lane & 15;
    const int n0 = wave * 32;        // first of two 16-wide n-tiles
    const int n1 = n0 + 16;

    // ---- cooperative async copy of the 16x256 bf16 A tile into LDS ----
    // 512 x 16B chunks; each of the 256 threads issues two async b128 loads.
    {
        unsigned loff = (unsigned)(size_t)(atile) + (unsigned)(tid * 16);
        unsigned long long gaddr =
            (unsigned long long)(size_t)(xln + (size_t)rowBase * C_DIM) + (unsigned long long)(tid * 16);
        asm volatile("global_load_async_to_lds_b128 %0, %1, off"
                     :: "v"(loff), "v"(gaddr) : "memory");
        asm volatile("global_load_async_to_lds_b128 %0, %1, off offset:4096"
                     :: "v"(loff), "v"(gaddr) : "memory");
    }
    // prefetch next tile's A rows while the async copy is in flight
    __builtin_prefetch(xln + (size_t)(rowBase + 16) * C_DIM + tid * 16, 0, 1);
    asm volatile("s_wait_asynccnt 0x0" ::: "memory");
    __syncthreads();

    // ---- GEMM 1: A from LDS, B = WgPack from global (L2-hot) ----
    const uint4* arow = (const uint4*)(atile + ln * C_DIM);
    v8f a0 = {0.f, 0.f, 0.f, 0.f, 0.f, 0.f, 0.f, 0.f};
    v8f a1 = {0.f, 0.f, 0.f, 0.f, 0.f, 0.f, 0.f, 0.f};
    #pragma unroll
    for (int kb = 0; kb < 8; ++kb) {
        TileBF A, B0, B1;
        // A lane layout (16-bit, 16x32): lane<16 -> K{0..7,16..23}, lane>=16 -> K{8..15,24..31}
        A.q[0] = arow[kb * 4 + hi];
        A.q[1] = arow[kb * 4 + hi + 2];
        const uint4* bp0 = (const uint4*)(WgP + (size_t)(kb * 256 + n0 + ln) * 32 + hi * 8);
        B0.q[0] = bp0[0]; B0.q[1] = bp0[2];
        const uint4* bp1 = (const uint4*)(WgP + (size_t)(kb * 256 + n1 + ln) * 32 + hi * 8);
        B1.q[0] = bp1[0]; B1.q[1] = bp1[2];
        a0 = __builtin_amdgcn_wmma_f32_16x16x32_bf16(false, A.v, false, B0.v, (short)0, a0, false, false);
        a1 = __builtin_amdgcn_wmma_f32_16x16x32_bf16(false, A.v, false, B1.v, (short)0, a1, false, false);
    }

    // ---- epilogue 1: +bg, sigmoid, * weighted[i], stash bf16 in LDS ----
    #pragma unroll
    for (int t2 = 0; t2 < 2; ++t2) {
        v8f acc = t2 ? a1 : a0;
        int n = (t2 ? n1 : n0) + ln;
        float bgn = bg[n];
        #pragma unroll
        for (int j = 0; j < 8; ++j) {
            int M = j + hi * 8;                       // row within tile
            float v = acc[j] + bgn;
            float g = 1.0f / (1.0f + __expf(-v));
            float gv = g * wvec[(size_t)(i0 + M) * 256 + n];
            gtile[M * LDS_PITCH + n] = f2bf(gv);
        }
    }
    __syncthreads();

    // ---- GEMM 2: gated(16xK from LDS) @ WoPack ----
    const uint4* grow = (const uint4*)(&gtile[ln * LDS_PITCH]);
    v8f o0 = {0.f, 0.f, 0.f, 0.f, 0.f, 0.f, 0.f, 0.f};
    v8f o1 = {0.f, 0.f, 0.f, 0.f, 0.f, 0.f, 0.f, 0.f};
    #pragma unroll
    for (int kb = 0; kb < 8; ++kb) {
        TileBF A, B0, B1;
        A.q[0] = grow[kb * 4 + hi];
        A.q[1] = grow[kb * 4 + hi + 2];
        const uint4* bp0 = (const uint4*)(WoP + (size_t)(kb * 256 + n0 + ln) * 32 + hi * 8);
        B0.q[0] = bp0[0]; B0.q[1] = bp0[2];
        const uint4* bp1 = (const uint4*)(WoP + (size_t)(kb * 256 + n1 + ln) * 32 + hi * 8);
        B1.q[0] = bp1[0]; B1.q[1] = bp1[2];
        o0 = __builtin_amdgcn_wmma_f32_16x16x32_bf16(false, A.v, false, B0.v, (short)0, o0, false, false);
        o1 = __builtin_amdgcn_wmma_f32_16x16x32_bf16(false, A.v, false, B1.v, (short)0, o1, false, false);
    }

    // ---- epilogue 2: +bo, *mask, store fp32 ----
    #pragma unroll
    for (int t2 = 0; t2 < 2; ++t2) {
        v8f acc = t2 ? o1 : o0;
        int n = (t2 ? n1 : n0) + ln;
        float bon = bo[n];
        #pragma unroll
        for (int j = 0; j < 8; ++j) {
            int M = j + hi * 8;
            float mk = mask[s * I_DIM + i0 + M];
            out[(size_t)(rowBase + M) * C_DIM + n] = (acc[j] + bon) * mk;
        }
    }
}

// ---- host launcher -------------------------------------------------------
extern "C" void kernel_launch(void* const* d_in, const int* in_sizes, int n_in,
                              void* d_out, int out_size, void* d_ws, size_t ws_size,
                              hipStream_t stream) {
    (void)in_sizes; (void)n_in; (void)out_size; (void)ws_size;
    const float* x    = (const float*)d_in[0];
    const float* mask = (const float*)d_in[1];
    const float* lnw  = (const float*)d_in[2];
    const float* lnb  = (const float*)d_in[3];
    const float* Wq   = (const float*)d_in[4];
    const float* Wk   = (const float*)d_in[5];
    const float* Wv   = (const float*)d_in[6];
    const float* Wg   = (const float*)d_in[7];
    const float* bg   = (const float*)d_in[8];
    const float* Wo   = (const float*)d_in[9];
    const float* bo   = (const float*)d_in[10];
    float* out = (float*)d_out;

    char* ws = (char*)d_ws;
    size_t off = 0;
    auto alloc = [&](size_t bytes) {
        void* p = ws + off;
        off += (bytes + 255) & ~(size_t)255;
        return p;
    };
    unsigned short* xln    = (unsigned short*)alloc((size_t)NROW * C_DIM * 2);        // 100.7 MB
    float*          scores = (float*)alloc((size_t)S_DIM * H_DIM * I_DIM * 4);        // 6.3 MB
    float*          attn   = (float*)alloc((size_t)S_DIM * H_DIM * I_DIM * 4);        // 6.3 MB
    float*          P      = (float*)alloc((size_t)I_DIM * H_DIM * C_DIM * 4);        // 3.1 MB
    float*          xm_ln  = (float*)alloc((size_t)I_DIM * C_DIM * 4);
    float*          wvec   = (float*)alloc((size_t)I_DIM * C_DIM * 4);
    unsigned short* WgP    = (unsigned short*)alloc((size_t)8 * 256 * 32 * 2);        // 128 KB
    unsigned short* WoP    = (unsigned short*)alloc((size_t)8 * 256 * 32 * 2);        // 128 KB

    k_pack<<<8, 256, 0, stream>>>(Wg, WgP);
    k_pack<<<8, 256, 0, stream>>>(Wo, WoP);
    k_mean<<<I_DIM, 256, 0, stream>>>(x, lnw, lnb, xm_ln);
    k_qp<<<I_DIM, 256, 0, stream>>>(xm_ln, Wq, Wk, P);
    k_ln_scores<<<NROW / 8, 256, 0, stream>>>(x, mask, lnw, lnb, P, xln, scores);
    k_softmax<<<H_DIM * I_DIM, 256, 0, stream>>>(scores, attn);
    k_accum<<<I_DIM, 256, 0, stream>>>(xln, attn, Wv, wvec);
    k_out<<<NROW / 16, 256, 0, stream>>>(xln, WgP, WoP, bg, bo, wvec, mask, out);
}
